// CrossEntropyLoss_38766374814398
// MI455X (gfx1250) — matrix operand order repaired and checked by reading, not compiled, not run on hardware
//
#include <hip/hip_runtime.h>

// Problem constants (B=2, S=2048, H=4096, V=32000)
constexpr int kT   = 4096;          // tokens = B*S
constexpr int kH   = 4096;          // hidden
constexpr int kV   = 32000;         // vocab
constexpr int kBM  = 128;           // token tile
constexpr int kBN  = 128;           // vocab tile
constexpr int kBK  = 32;            // K chunk (= WMMA K for bf16)
constexpr int kNCH = kV / kBN;      // 250 vocab chunks

typedef __attribute__((ext_vector_type(16))) __bf16 v16bf;
typedef __attribute__((ext_vector_type(8)))  float  v8f;
typedef unsigned int u32x4 __attribute__((ext_vector_type(4)));
typedef int          i32x8 __attribute__((ext_vector_type(8)));
typedef int          i32x4 __attribute__((ext_vector_type(4)));

#if defined(__has_builtin)
#if __has_builtin(__builtin_amdgcn_tensor_load_to_lds) && __has_builtin(__builtin_amdgcn_s_wait_tensorcnt)
#define HAS_TDM 1
#endif
#endif
#ifndef HAS_TDM
#define HAS_TDM 0
#endif

// ---------------------------------------------------------------- f32 -> bf16
__device__ __forceinline__ unsigned short f2bf(float f) {
    unsigned u = __float_as_uint(f);
    u += 0x7FFFu + ((u >> 16) & 1u);       // round-to-nearest-even
    return (unsigned short)(u >> 16);
}

__global__ void ce_cvt_bf16(const float* __restrict__ src,
                            unsigned short* __restrict__ dst, long long n4) {
    long long i = (long long)blockIdx.x * blockDim.x + threadIdx.x;
    long long stride = (long long)gridDim.x * blockDim.x;
    for (; i < n4; i += stride) {
        float4 f = ((const float4*)src)[i];
        ushort4 o;
        o.x = f2bf(f.x); o.y = f2bf(f.y); o.z = f2bf(f.z); o.w = f2bf(f.w);
        ((ushort4*)dst)[i] = o;
    }
}

#if HAS_TDM
// Issue one TDM 2D tile load: kBK elems (2B) x kBM rows, row stride kH elems.
// gptr already points at the tile start (row0 * kH + k0).
__device__ __forceinline__ void tdm_load_tile(unsigned lds_off,
                                              const unsigned short* gptr,
                                              unsigned rows_total) {
    unsigned long long ga = (unsigned long long)gptr;
    u32x4 g0;
    g0.x = 1u;                                            // count=1, user-mode
    g0.y = lds_off;                                       // LDS byte address
    g0.z = (unsigned)(ga & 0xFFFFFFFFull);                // global_addr[31:0]
    g0.w = (unsigned)((ga >> 32) & 0x1FFFFFFull) | (2u << 30); // [56:32] + type=2
    i32x8 g1;
    g1[0] = 0x00010000;                                   // data_size = 2 bytes
    g1[1] = (int)((kH & 0xFFFF) << 16);                   // tensor_dim0[15:0]
    g1[2] = (int)(((kH >> 16) & 0xFFFF) | ((rows_total & 0xFFFF) << 16));
    g1[3] = (int)(((rows_total >> 16) & 0xFFFF) | (kBK << 16)); // tile_dim0=32
    g1[4] = kBM;                                          // tile_dim1 = 128
    g1[5] = kH;                                           // tensor_dim0_stride
    g1[6] = 0;
    g1[7] = 0;
    i32x4 gz4 = {0, 0, 0, 0};
    i32x8 gz8 = {0, 0, 0, 0, 0, 0, 0, 0};
    // 6-arg form (this toolchain): groups 0..3 + extra group + cache policy
    __builtin_amdgcn_tensor_load_to_lds(g0, g1, gz4, gz4, gz8, 0);
}
#endif

// ------------------------------------------- fused GEMM + partial log-softmax
// grid = (kT/kBM = 32 [fast], kNCH = 250), block = 256 (8 waves)
__global__ __launch_bounds__(256)
void ce_gemm_partial(const unsigned short* __restrict__ hB,   // [kT][kH] bf16 bits
                     const unsigned short* __restrict__ wB,   // [kV][kH] bf16 bits
                     const int* __restrict__ labels,          // [kT]
                     float* __restrict__ pMax, float* __restrict__ pSum,
                     float* __restrict__ pPick) {
    __shared__ unsigned short As[2][kBM * kBK];   // 2 x 8 KB
    __shared__ unsigned short Bs[2][kBN * kBK];   // 2 x 8 KB

    const int tid   = threadIdx.x;
    const int lane  = tid & 31;
    const int wv    = tid >> 5;
    const int half  = lane >> 4;
    const int lrow  = lane & 15;
    const int tileM = blockIdx.x * kBM;
    const int chunk = blockIdx.y;
    const int tileN = chunk * kBN;

    v8f acc[8];
    #pragma unroll
    for (int c = 0; c < 8; ++c) acc[c] = (v8f){0.f,0.f,0.f,0.f,0.f,0.f,0.f,0.f};

    // One K-step of WMMA work on the given LDS tiles, B frags software-pipelined.
    auto compute = [&](const unsigned short* Ac, const unsigned short* Bc) {
        union FU { int4 q[2]; v16bf v; };
        FU a;
        const unsigned short* arow = Ac + (wv * 16 + lrow) * kBK + half * 8;
        a.q[0] = *(const int4*)(arow);          // K = kb .. kb+7
        a.q[1] = *(const int4*)(arow + 16);     // K = 16+kb ..
        FU bf[2];
        {
            const unsigned short* brow = Bc + lrow * kBK + half * 8;
            bf[0].q[0] = *(const int4*)(brow);
            bf[0].q[1] = *(const int4*)(brow + 16);
        }
        #pragma unroll
        for (int c = 0; c < 8; ++c) {
            if (c + 1 < 8) {                    // prefetch next B frag into other slot
                const unsigned short* brow = Bc + ((c + 1) * 16 + lrow) * kBK + half * 8;
                bf[(c + 1) & 1].q[0] = *(const int4*)(brow);
                bf[(c + 1) & 1].q[1] = *(const int4*)(brow + 16);
            }
            acc[c] = __builtin_amdgcn_wmma_f32_16x16x32_bf16(
                false, a.v, false, bf[c & 1].v, (short)0, acc[c], false, false);
        }
    };

#if HAS_TDM
    // ---- Tensor Data Mover double-buffered pipeline, unrolled x2 so each
    // phase uses compile-time-constant LDS buffers (no acc register rotation).
    const int nk = kH / kBK;                    // 128 K-steps (even)
    const unsigned ldsA0 = (unsigned)(size_t)&As[0][0];
    const unsigned ldsA1 = (unsigned)(size_t)&As[1][0];
    const unsigned ldsB0 = (unsigned)(size_t)&Bs[0][0];
    const unsigned ldsB1 = (unsigned)(size_t)&Bs[1][0];
    const unsigned short* gA = hB + (size_t)tileM * kH;
    const unsigned short* gB = wB + (size_t)tileN * kH;

    if (wv == 0) {
        tdm_load_tile(ldsA0, gA, kT);
        tdm_load_tile(ldsB0, gB, kV);
        __builtin_amdgcn_s_wait_tensorcnt(0);
    }
    __syncthreads();
    for (int kk = 0; kk < nk; kk += 2) {
        // phase A: compute buf0 while TDM fills buf1 with K-step kk+1
        if (wv == 0) {
            int k1 = (kk + 1) * kBK;
            tdm_load_tile(ldsA1, gA + k1, kT);
            tdm_load_tile(ldsB1, gB + k1, kV);
        }
        compute(As[0], Bs[0]);
        if (wv == 0) __builtin_amdgcn_s_wait_tensorcnt(0);
        __syncthreads();

        // phase B: compute buf1 while TDM fills buf0 with K-step kk+2
        bool more = (kk + 2 < nk);
        if (wv == 0 && more) {
            int k2 = (kk + 2) * kBK;
            tdm_load_tile(ldsA0, gA + k2, kT);
            tdm_load_tile(ldsB0, gB + k2, kV);
        }
        compute(As[1], Bs[1]);
        if (wv == 0 && more) __builtin_amdgcn_s_wait_tensorcnt(0);
        __syncthreads();
    }
#else
    // ---- fallback: cooperative VGPR staging (global_load_b128 -> ds_store_b128)
    for (int k0 = 0; k0 < kH; k0 += kBK) {
        #pragma unroll
        for (int i = 0; i < 2; ++i) {
            int chnk = tid + i * 256;          // 0..511
            int row  = chnk >> 2;              // 0..127
            int qcol = chnk & 3;               // which 16B of the 64B row
            const uint4* ga = (const uint4*)(hB + (size_t)(tileM + row) * kH + k0) + qcol;
            const uint4* gb = (const uint4*)(wB + (size_t)(tileN + row) * kH + k0) + qcol;
            ((uint4*)(&As[0][row * kBK]))[qcol] = *ga;
            ((uint4*)(&Bs[0][row * kBK]))[qcol] = *gb;
            if (k0 + kBK < kH) {
                __builtin_prefetch((const void*)(ga + kBK / 8), 0, 0);
                __builtin_prefetch((const void*)(gb + kBK / 8), 0, 0);
            }
        }
        __syncthreads();
        compute(As[0], Bs[0]);
        __syncthreads();
    }
#endif

    // ---- epilogue: per-row (token) max / sum-exp / picked over this 128-col chunk
    // C layout: lanes 0-15 hold rows 0..7 (reg r), cols = lane; lanes 16-31 rows 8..15
    float rmax[8], rsum[8], rpick[8];
    #pragma unroll
    for (int r = 0; r < 8; ++r) {
        float m = acc[0][r];
        #pragma unroll
        for (int c = 1; c < 8; ++c) m = fmaxf(m, acc[c][r]);
        #pragma unroll
        for (int off = 8; off >= 1; off >>= 1) m = fmaxf(m, __shfl_xor(m, off, 32));
        rmax[r] = m;

        float s = 0.f;
        #pragma unroll
        for (int c = 0; c < 8; ++c) s += __expf(acc[c][r] - m);
        #pragma unroll
        for (int off = 8; off >= 1; off >>= 1) s += __shfl_xor(s, off, 32);
        rsum[r] = s;

        int row = half * 8 + r;
        int tok = tileM + wv * 16 + row;
        int lbl = labels[tok];
        int j   = lbl - tileN;
        bool inr = (j >= 0) && (j < kBN);
        int jc   = inr ? j : 0;
        int cblk = jc >> 4, within = jc & 15;
        float sel = acc[0][r];
        #pragma unroll
        for (int c = 1; c < 8; ++c) sel = (cblk == c) ? acc[c][r] : sel;
        float pv = __shfl(sel, half * 16 + within, 32);
        rpick[r] = inr ? pv : 0.f;
    }

    if (lrow == 0) {                 // lanes 0 (rows 0-7) and 16 (rows 8-15)
        #pragma unroll
        for (int r = 0; r < 8; ++r) {
            int tok = tileM + wv * 16 + half * 8 + r;
            size_t idx = (size_t)tok * kNCH + chunk;
            pMax[idx]  = rmax[r];
            pSum[idx]  = rsum[r];
            pPick[idx] = rpick[r];
        }
    }
}

// -------------------------------------------------- split-softmax combine
__global__ void ce_zero(float* accum) { accum[0] = 0.f; accum[1] = 0.f; }

__global__ void ce_reduce(const float* __restrict__ pMax,
                          const float* __restrict__ pSum,
                          const float* __restrict__ pPick,
                          const float* __restrict__ lw,
                          float* accum) {
    int t = blockIdx.x * blockDim.x + threadIdx.x;
    float num = 0.f, den = 0.f;
    if (t < kT) {
        const size_t base = (size_t)t * kNCH;
        float M = -3.4e38f;
        for (int ch = 0; ch < kNCH; ++ch) M = fmaxf(M, pMax[base + ch]);
        float Z = 0.f, P = 0.f;
        for (int ch = 0; ch < kNCH; ++ch) {
            Z += pSum[base + ch] * __expf(pMax[base + ch] - M);
            P += pPick[base + ch];
        }
        float per = M + __logf(Z) - P;   // logsumexp - picked
        float w   = lw[t];
        num = per * w;
        den = w;
    }
    __shared__ float s1[256], s2[256];
    s1[threadIdx.x] = num; s2[threadIdx.x] = den;
    __syncthreads();
    for (int s = 128; s > 0; s >>= 1) {
        if (threadIdx.x < s) {
            s1[threadIdx.x] += s1[threadIdx.x + s];
            s2[threadIdx.x] += s2[threadIdx.x + s];
        }
        __syncthreads();
    }
    if (threadIdx.x == 0) {
        atomicAdd(&accum[0], s1[0]);
        atomicAdd(&accum[1], s2[0]);
    }
}

__global__ void ce_final(const float* accum, const int* gas, float* out) {
    out[0] = accum[0] / (accum[1] + 1e-8f) / (float)gas[0];
}

// ------------------------------------------------------------------ launcher
extern "C" void kernel_launch(void* const* d_in, const int* in_sizes, int n_in,
                              void* d_out, int out_size, void* d_ws, size_t ws_size,
                              hipStream_t stream) {
    const float* hidden = (const float*)d_in[0];   // [kT][kH]
    const float* weight = (const float*)d_in[1];   // [kV][kH]
    const int*   labels = (const int*)d_in[2];     // [kT]
    const float* lw     = (const float*)d_in[3];   // [kT]
    const int*   gas    = (const int*)d_in[4];     // [1]
    float* out = (float*)d_out;

    auto align256 = [](size_t x) { return (x + 255) & ~(size_t)255; };
    char* ws = (char*)d_ws;
    size_t off = 0;
    unsigned short* hB = (unsigned short*)(ws + off); off = align256(off + (size_t)kT * kH * 2);
    unsigned short* wB = (unsigned short*)(ws + off); off = align256(off + (size_t)kV * kH * 2);
    float* pMax  = (float*)(ws + off); off = align256(off + (size_t)kT * kNCH * 4);
    float* pSum  = (float*)(ws + off); off = align256(off + (size_t)kT * kNCH * 4);
    float* pPick = (float*)(ws + off); off = align256(off + (size_t)kT * kNCH * 4);
    float* accum = (float*)(ws + off); off = align256(off + 256);
    (void)ws_size; (void)in_sizes; (void)n_in; (void)out_size;

    // 1) f32 -> bf16 conversions
    ce_cvt_bf16<<<2048, 256, 0, stream>>>(hidden, hB, (long long)kT * kH / 4);
    ce_cvt_bf16<<<8192, 256, 0, stream>>>(weight, wB, (long long)kV * kH / 4);

    // 2) fused WMMA GEMM + per-chunk softmax partials
    dim3 grid(kT / kBM, kNCH);            // token tiles fast => weight tile L2-resident
    ce_gemm_partial<<<grid, 256, 0, stream>>>(hB, wB, labels, pMax, pSum, pPick);

    // 3) combine partials -> weighted scalar loss
    ce_zero<<<1, 1, 0, stream>>>(accum);
    ce_reduce<<<kT / 256, 256, 0, stream>>>(pMax, pSum, pPick, lw, accum);
    ce_final<<<1, 1, 0, stream>>>(accum, gas, out);
}
